// DigitConvolutionalModel_6880537608268
// MI455X (gfx1250) — matrix operand order repaired and checked by reading, not compiled, not run on hardware
//
#include <hip/hip_runtime.h>

// ---- types --------------------------------------------------------------
typedef __attribute__((ext_vector_type(16))) __bf16 v16bf;
typedef __attribute__((ext_vector_type(8)))  __bf16 v8bf;
typedef __attribute__((ext_vector_type(8)))  float   v8f;
typedef __attribute__((ext_vector_type(4)))  int     v4i;
typedef unsigned short bf16_raw;   // host-visible handle for bf16 buffers

#define IMG_H 28
#define IMG_W 28
#define CONV_OUT 676          // 26*26
#define KPAD 768              // 6 * 128 = 24 * 32
#define CHUNK_K 128
#define NCHUNK 6
#define HIDDEN 128
#define OUTN 10
#define TILE_M 128
#define FEAT_STRIDE 776       // bf16; 776*2B = 388 dw == 4 (mod 64) -> conflict-free A reads
#define W1LDS_STRIDE 136      // bf16; 68 dw == 4 (mod 64) -> conflict-free B reads
#define H_STRIDE 136
#define W2_STRIDE 136

#if defined(__HIP_DEVICE_COMPILE__) && __has_builtin(__builtin_amdgcn_global_load_async_to_lds_b128)
#define ASYNC_OK 1
typedef __attribute__((address_space(1))) v4i v4i_g;   // global (device) AS
typedef __attribute__((address_space(3))) v4i v4i_l;   // LDS (shared) AS
#else
#define ASYNC_OK 0
#endif

__device__ __forceinline__ v16bf cat8(v8bf lo, v8bf hi) {
  return __builtin_shufflevector(lo, hi, 0,1,2,3,4,5,6,7,8,9,10,11,12,13,14,15);
}

__device__ __forceinline__ void wait_async_lds() {
#if ASYNC_OK
 #if __has_builtin(__builtin_amdgcn_s_wait_asynccnt)
  __builtin_amdgcn_s_wait_asynccnt(0);
 #else
  asm volatile("s_wait_asynccnt 0" ::: "memory");
 #endif
#endif
}

// Stage one 128(N) x 128(K) bf16 chunk of w1 into LDS (stride W1LDS_STRIDE).
// 2048 x 16B transfers, 8 per thread, ASYNCcnt-tracked when available.
__device__ __forceinline__ void stage_w1_chunk(const __bf16* __restrict__ w1bf,
                                               __bf16* __restrict__ wbuf,
                                               int chunk, int tid) {
  #pragma unroll
  for (int i = 0; i < 8; ++i) {
    int e  = tid + (i << 8);        // 0..2047
    int n  = e >> 4;                // row (N)
    int kq = e & 15;                // which 16-byte segment of the 256B row
    const __bf16* g = w1bf + (size_t)n * KPAD + (chunk << 7) + (kq << 3);
    __bf16* l = wbuf + n * W1LDS_STRIDE + (kq << 3);
#if ASYNC_OK
    __builtin_amdgcn_global_load_async_to_lds_b128((v4i_g*)g, (v4i_l*)l, 0, 0);
#else
    *(v8bf*)l = *(const v8bf*)g;    // fallback: global_load_b128 + ds_store_b128
#endif
  }
}

// ---- prep: cast + pad weights to bf16 in workspace ----------------------
extern "C" __global__ void __launch_bounds__(256)
prep_weights_k(const float* __restrict__ w1, const float* __restrict__ w2,
               bf16_raw* __restrict__ w1bf_raw, bf16_raw* __restrict__ w2bf_raw) {
  __bf16* w1bf = reinterpret_cast<__bf16*>(w1bf_raw);
  __bf16* w2bf = reinterpret_cast<__bf16*>(w2bf_raw);
  int idx = blockIdx.x * 256 + threadIdx.x;
  const int N1 = HIDDEN * KPAD;                 // 98304
  if (idx < N1) {
    int r = idx / KPAD, k = idx - r * KPAD;
    float v = (k < CONV_OUT) ? w1[r * CONV_OUT + k] : 0.0f;
    w1bf[idx] = (__bf16)v;
  } else {
    int j = idx - N1;                           // padded w2: [16][128]
    if (j < 16 * HIDDEN) {
      int r = j >> 7, k = j & 127;
      float v = (r < OUTN) ? w2[r * HIDDEN + k] : 0.0f;
      w2bf[j] = (__bf16)v;
    }
  }
}

// ---- fused conv + FC1(relu) + FC2 ---------------------------------------
extern "C" __global__ void __launch_bounds__(256)
fused_mlp_k(const float* __restrict__ x, const float* __restrict__ cw,
            const bf16_raw* __restrict__ w1bf_raw, const bf16_raw* __restrict__ w2bf_raw,
            const float* __restrict__ b1, const float* __restrict__ b2,
            float* __restrict__ out) {
  const __bf16* w1bf = reinterpret_cast<const __bf16*>(w1bf_raw);
  const __bf16* w2bf = reinterpret_cast<const __bf16*>(w2bf_raw);

  extern __shared__ char smem[];
  __bf16* featL = (__bf16*)smem;                         // 128 x 776       198,656 B
  __bf16* wbuf0 = featL + TILE_M * FEAT_STRIDE;          // 128 x 136        34,816 B
  __bf16* wbuf1 = wbuf0 + HIDDEN * W1LDS_STRIDE;         // 128 x 136        34,816 B
  __bf16* hL    = wbuf1 + HIDDEN * W1LDS_STRIDE;         // 128 x 136        34,816 B
  __bf16* w2L   = hL    + TILE_M * H_STRIDE;             // 16  x 136         4,352 B
  float*  b1L   = (float*)(w2L + 16 * W2_STRIDE);        // 128                 512 B
  float*  b2L   = b1L + HIDDEN;                          // 16                   64 B

  const int tid  = threadIdx.x;
  const int lane = tid & 31;
  const int wid  = tid >> 5;                 // 0..7 (8 wave32s)
  const int blockRow = blockIdx.x * TILE_M;

  // kick off async staging of the first w1 chunk immediately
  stage_w1_chunk(w1bf, wbuf0, 0, tid);

  // stage biases + padded w2 into LDS
  if (tid < HIDDEN) b1L[tid] = b1[tid];
  if (tid < 16)     b2L[tid] = (tid < OUTN) ? b2[tid] : 0.0f;
  for (int i = tid; i < 16 * HIDDEN; i += 256) {
    int r = i >> 7, k = i & 127;
    w2L[r * W2_STRIDE + k] = w2bf[i];
  }

  // conv weights: uniform -> scalar loads
  const float c00=cw[0],c01=cw[1],c02=cw[2],c10=cw[3],c11=cw[4],
              c12=cw[5],c20=cw[6],c21=cw[7],c22=cw[8];

  // phase 1: conv (f32 VALU) + cast to bf16 feature tile in LDS, K padded w/ 0
  for (int idx = tid; idx < TILE_M * KPAD; idx += 256) {
    int r = idx / KPAD;
    int j = idx - r * KPAD;
    float v = 0.0f;
    if (j < CONV_OUT) {
      int oy = j / 26, ox = j - oy * 26;
      const float* xr = x + (size_t)(blockRow + r) * (IMG_H * IMG_W) + oy * IMG_W + ox;
      v = xr[0]  * c00 + xr[1]  * c01 + xr[2]  * c02
        + xr[28] * c10 + xr[29] * c11 + xr[30] * c12
        + xr[56] * c20 + xr[57] * c21 + xr[58] * c22;
    }
    featL[r * FEAT_STRIDE + j] = (__bf16)v;
  }
  wait_async_lds();          // our chunk-0 transfers have landed
  __syncthreads();           // everyone's conv tile + chunk 0 + w2/bias visible

  // phase 2: FC1 -- each wave owns 16 rows x 128 cols (8 WMMA tiles along N)
  const int mrow = (wid << 4) + (lane & 15);          // local batch row for A frag
  const int ncol = lane & 15;                         // N within a 16-wide tile
  const int offA = (lane >= 16) ? 8 : 0;              // 16-bit A-frag K split
  const int offB = (lane >= 16) ? 8 : 0;              // 16-bit B-frag K split
  const int hiM  = (lane >= 16) ? 8 : 0;              // C/D row offset

  v8f acc[8];
  #pragma unroll
  for (int nt = 0; nt < 8; ++nt) { v8f z = {}; acc[nt] = z; }

  for (int c = 0; c < NCHUNK; ++c) {
    __bf16* cur = (c & 1) ? wbuf1 : wbuf0;
    __bf16* nxt = (c & 1) ? wbuf0 : wbuf1;
    if (c + 1 < NCHUNK)                       // overlap next-chunk DMA with math
      stage_w1_chunk(w1bf, nxt, c + 1, tid);

    #pragma unroll
    for (int kk = 0; kk < CHUNK_K; kk += 32) {
      const int kc = (c << 7) + kk;
      const __bf16* ap = featL + mrow * FEAT_STRIDE + kc + offA;
      v16bf a = cat8(*(const v8bf*)ap, *(const v8bf*)(ap + 16));
      #pragma unroll
      for (int nt = 0; nt < 8; ++nt) {
        const __bf16* bp = cur + ((nt << 4) + ncol) * W1LDS_STRIDE + kk + offB;
        v16bf b = cat8(*(const v8bf*)bp, *(const v8bf*)(bp + 16));
        acc[nt] = __builtin_amdgcn_wmma_f32_16x16x32_bf16(
            false, a, false, b, (short)0, acc[nt], false, false);
      }
    }
    wait_async_lds();        // our async writes into `nxt` are done
    __syncthreads();         // all waves done reading `cur`, `nxt` visible
  }

  // bias + relu, write h tile as bf16 (C/D layout: lane%16 = N, vgpr r = row)
  #pragma unroll
  for (int nt = 0; nt < 8; ++nt) {
    float bias = b1L[(nt << 4) + ncol];
    #pragma unroll
    for (int r = 0; r < 8; ++r) {
      float hv = acc[nt][r] + bias;
      hv = hv > 0.0f ? hv : 0.0f;
      hL[((wid << 4) + r + hiM) * H_STRIDE + (nt << 4) + ncol] = (__bf16)hv;
    }
  }
  __syncthreads();

  // phase 3: FC2 -- 16x16 tile per wave (N padded 10->16), K = 128 = 4 steps
  v8f acc2 = {};
  #pragma unroll
  for (int kc = 0; kc < HIDDEN; kc += 32) {
    const __bf16* ap = hL + ((wid << 4) + (lane & 15)) * H_STRIDE + kc + offA;
    v16bf a = cat8(*(const v8bf*)ap, *(const v8bf*)(ap + 16));
    const __bf16* bp = w2L + ncol * W2_STRIDE + kc + offB;
    v16bf b = cat8(*(const v8bf*)bp, *(const v8bf*)(bp + 16));
    acc2 = __builtin_amdgcn_wmma_f32_16x16x32_bf16(
        false, a, false, b, (short)0, acc2, false, false);
  }

  if (ncol < OUTN) {
    float bias2 = b2L[ncol];
    #pragma unroll
    for (int r = 0; r < 8; ++r) {
      int m = (wid << 4) + r + hiM;
      out[(size_t)(blockRow + m) * OUTN + ncol] = acc2[r] + bias2;
    }
  }
}

// ---- launcher -----------------------------------------------------------
extern "C" void kernel_launch(void* const* d_in, const int* in_sizes, int n_in,
                              void* d_out, int out_size, void* d_ws, size_t ws_size,
                              hipStream_t stream) {
  const float* x  = (const float*)d_in[0];
  const float* cw = (const float*)d_in[1];
  const float* w1 = (const float*)d_in[2];
  const float* b1 = (const float*)d_in[3];
  const float* w2 = (const float*)d_in[4];
  const float* b2 = (const float*)d_in[5];
  float* out = (float*)d_out;

  bf16_raw* w1bf = (bf16_raw*)d_ws;                       // 128*768 bf16
  bf16_raw* w2bf = w1bf + HIDDEN * KPAD;                  // 16*128 bf16

  const int B = in_sizes[0] / (IMG_H * IMG_W);            // 65536

  const int prepElems = HIDDEN * KPAD + 16 * HIDDEN;      // 100352
  prep_weights_k<<<(prepElems + 255) / 256, 256, 0, stream>>>(w1, w2, w1bf, w2bf);

  size_t lds = (size_t)(TILE_M * FEAT_STRIDE            // feat tile
                        + 2 * HIDDEN * W1LDS_STRIDE     // w1 double buffer
                        + TILE_M * H_STRIDE             // h tile
                        + 16 * W2_STRIDE) * 2           // padded w2
             + (HIDDEN + 16) * sizeof(float);           // biases -> 308,032 B
  fused_mlp_k<<<B / TILE_M, 256, lds, stream>>>(x, cw, w1bf, w2bf, b1, b2, out);
}
